// MS_Attention_linear_82076825027347
// MI455X (gfx1250) — matrix operand order repaired and checked by reading, not compiled, not run on hardware
//
#include <hip/hip_runtime.h>

// MS_Attention_linear for MI455X (gfx1250, wave32, WMMA).
// Linear-attention refactor: o = q @ (k^T @ v) * 0.25 (no softmax in reference).
// All matmuls use v_wmma_f32_16x16x32_f16; activations are spiked to multiples
// of 0.25 <= 1.0, which are exact in fp16. GEMM W-tiles are staged to LDS via
// async global->LDS copies (ASYNCcnt), double-buffered with in-flight waits.

#define BB     8
#define CC     512
#define NN     1024   // H*W
#define NHEADS 8
#define HD     64

typedef __attribute__((ext_vector_type(8)))  _Float16 v8h;
typedef __attribute__((ext_vector_type(16))) _Float16 v16h;
typedef __attribute__((ext_vector_type(8)))  float    v8f;
typedef __attribute__((ext_vector_type(4)))  int      v4i;

#if defined(__has_builtin)
# if __has_builtin(__builtin_amdgcn_global_load_async_to_lds_b128)
#  define HAVE_ASYNC_LDS 1
# endif
#endif
#ifndef HAVE_ASYNC_LDS
# define HAVE_ASYNC_LDS 0
#endif

typedef __attribute__((address_space(1))) v4i gv4i;   // global v4i
typedef __attribute__((address_space(3))) v4i lv4i;   // LDS v4i

#if HAVE_ASYNC_LDS
# if __has_builtin(__builtin_amdgcn_s_wait_asynccnt)
#  define WAIT_ASYNC(n) __builtin_amdgcn_s_wait_asynccnt(n)
# else
#  define WAIT_ASYNC(n) asm volatile("s_wait_asynccnt %0" :: "i"(n) : "memory")
# endif
#else
# define WAIT_ASYNC(n) ((void)0)
#endif

__device__ __forceinline__ float spikef(float x) {
    x = fminf(fmaxf(x, 0.0f), 4.0f);
    return rintf(x) * 0.25f;              // round(clamp(x,0,4))/4
}

// 16-byte global -> LDS copy; async (ASYNCcnt) when available.
__device__ __forceinline__ void async_cp_b128(const _Float16* g, _Float16* l) {
#if HAVE_ASYNC_LDS
    gv4i* gp = (gv4i*)(v4i*)g;            // generic -> AS1 (addrspacecast)
    lv4i* lp = (lv4i*)(v4i*)l;            // generic -> AS3 (addrspacecast)
    __builtin_amdgcn_global_load_async_to_lds_b128(gp, lp, 0, 0);
#else
    *(v8h*)l = *(const v8h*)g;
#endif
}

// ---- WMMA fragment loaders (ISA 7.12.2 layouts, wave32) -------------------
// A fragment: 16x32 f16 (rows x K), source row-major, ld = row stride (halves).
// Lane l: row = l&15, grp = l>>4; half i holds K = (i<8?0:16) + 8*grp + (i&7).
__device__ __forceinline__ v16h load_frag_a(const _Float16* __restrict__ p0,
                                            int ld, int lane) {
    const int row = lane & 15;
    const int grp = lane >> 4;
    const _Float16* p = p0 + (size_t)row * ld + grp * 8;
    union { v16h f; struct { v8h lo; v8h hi; } h; } u;
    u.h.lo = *(const v8h*)(p);            // K = 8*grp + 0..7
    u.h.hi = *(const v8h*)(p + 16);       // K = 16 + 8*grp + 0..7
    return u.f;
}

// B fragment: 32x16 f16 (K x cols), source holds B^T row-major (row = col),
// ld = per-column stride (halves). Lane l: col = l&15, K = (i<8?0:16) + 8*(l>>4) + (i&7).
__device__ __forceinline__ v16h load_frag_bT(const _Float16* __restrict__ p0,
                                             int ld, int lane) {
    const int col = lane & 15;
    const int grp = lane >> 4;
    const _Float16* p = p0 + (size_t)col * ld + grp * 8;
    union { v16h f; struct { v8h lo; v8h hi; } h; } u;
    u.h.lo = *(const v8h*)(p);
    u.h.hi = *(const v8h*)(p + 16);
    return u.f;
}

__device__ __forceinline__ v8f wmma16(v16h a, v16h b, v8f c) {
    return __builtin_amdgcn_wmma_f32_16x16x32_f16(
        /*neg_a=*/false, a, /*neg_b=*/false, b,
        /*c_mod=*/(short)0, c, /*reuse_a=*/false, /*reuse_b=*/false);
}

// ---- K0a: fp32 weights -> fp16 -------------------------------------------
__global__ void wconv_kernel(const float* __restrict__ w,
                             _Float16* __restrict__ wh, int n) {
    int i = blockIdx.x * blockDim.x + threadIdx.x;
    if (i < n) wh[i] = (_Float16)w[i];
}

// ---- K0b: eval-BN folding: s = g*rsqrt(v+eps), t = b - m*s ---------------
__global__ void bnprep_kernel(const float* __restrict__ g, const float* __restrict__ b,
                              const float* __restrict__ m, const float* __restrict__ v,
                              float* __restrict__ s, float* __restrict__ t) {
    int i = blockIdx.x * blockDim.x + threadIdx.x;
    if (i < CC) {
        float sc = g[i] * rsqrtf(v[i] + 1e-5f);
        s[i] = sc;
        t[i] = b[i] - m[i] * sc;
    }
}

// ---- K1: xs = spike(x), transposed to [b][n][c] fp16 (LDS-tiled) ---------
__global__ __launch_bounds__(256) void spike_x_kernel(const float* __restrict__ x,
                                                      _Float16* __restrict__ xT) {
    __shared__ _Float16 tile[32][33];
    const int b  = blockIdx.z;
    const int c0 = blockIdx.y * 32;
    const int n0 = blockIdx.x * 32;
    const int tx = threadIdx.x, ty = threadIdx.y;     // 32 x 8
#pragma unroll
    for (int i = 0; i < 4; ++i) {
        int c = c0 + ty + i * 8;
        float val = x[((size_t)b * CC + c) * NN + n0 + tx];
        tile[ty + i * 8][tx] = (_Float16)spikef(val);
    }
    __syncthreads();
#pragma unroll
    for (int i = 0; i < 4; ++i) {
        int n = n0 + ty + i * 8;
        xT[((size_t)b * NN + n) * CC + c0 + tx] = tile[tx][ty + i * 8];
    }
}

// ---- K2: Y[o,(b,n)] = W[o,:] . X[:, (b,n)]  + BN (+spike) ----------------
// A = Wh [512x512], staged per 128x32 tile into LDS (async, ping-pong bufs);
// B from Xt [b][n][c] (== B^T row-major), direct-global, ping-pong registers.
// k-loop unrolled by 2 so loads write their destination set directly (no
// register-copy rotation). Block = 128x128 tile; 8 waves, 64x32 each.
// MODE 0: spike -> f16 out [b][n][c]   (for q)
// MODE 1: spike -> f16 out [b][c][n]   (for k, v)
// MODE 2: raw   -> f32 out [b][c][n]   (final projection -> d_out)
template<int MODE>
__global__ __launch_bounds__(256) void gemm_bn_kernel(
    const _Float16* __restrict__ Wh, const _Float16* __restrict__ Xt,
    const float* __restrict__ bnS, const float* __restrict__ bnT,
    void* __restrict__ outp)
{
    __shared__ __align__(16) _Float16 wtile[2][128 * 32];   // 16 KB ping-pong

    const int lane  = threadIdx.x & 31;
    const int wave  = threadIdx.x >> 5;
    const int col0  = blockIdx.x * 128 + (wave & 3) * 32;   // flattened (b,n)
    const int o0blk = blockIdx.y * 128;                     // block row base
    const int oW    = (wave >> 2) * 64;                     // wave row base in tile
    const int o0    = o0blk + oW;                           // wave row base global
    const int b     = col0 / NN;
    const int n0    = col0 - b * NN;                        // 128-col tile stays in one b
    const _Float16* __restrict__ xb   = Xt + ((size_t)b * NN + n0) * CC;
    const _Float16* __restrict__ wblk = Wh + (size_t)o0blk * CC;

    // Cooperative W-tile stage: 128 rows x 32 halves = 8 KB, 2 async b128/thread
    // (= wave ASYNCcnt += 2 per stage; async loads complete in order).
    const int r0 = threadIdx.x >> 2, s0 = threadIdx.x & 3;
    const int r1 = (threadIdx.x + 256) >> 2, s1 = (threadIdx.x + 256) & 3;
    auto stage = [&](int k, int buf) {
        async_cp_b128(wblk + (size_t)r0 * CC + k + s0 * 8, &wtile[buf][r0 * 32 + s0 * 8]);
        async_cp_b128(wblk + (size_t)r1 * CC + k + s1 * 8, &wtile[buf][r1 * 32 + s1 * 8]);
    };

    v8f acc[4][2] = {};
    auto loadB = [&](v16h* dst, int k) {
#pragma unroll
        for (int c = 0; c < 2; ++c)
            dst[c] = load_frag_bT(xb + (size_t)(c * 16) * CC + k, CC, lane);
    };
    auto mma = [&](const v16h* bfr, int buf) {
        v16h afr[4];
#pragma unroll
        for (int r = 0; r < 4; ++r)                 // ds_load_b128 x2 per frag
            afr[r] = load_frag_a(&wtile[buf][(oW + r * 16) * 32], 32, lane);
#pragma unroll
        for (int r = 0; r < 4; ++r)
#pragma unroll
            for (int c = 0; c < 2; ++c)
                acc[r][c] = wmma16(afr[r], bfr[c], acc[r][c]);
    };

    stage(0, 0);                                    // prime both LDS buffers
    stage(32, 1);
    v16h bA[2], bB[2];
    loadB(bA, 0);

    for (int kk = 0; kk < CC; kk += 64) {
        // ---- step k = kk: consume wtile[0] + bA --------------------------
        WAIT_ASYNC(2);                              // oldest stage (buf0) landed
        __syncthreads();
        loadB(bB, kk + 32);                         // kk+32 < CC always
        __builtin_prefetch(xb + kk + 96, 0, 0);
        mma(bA, 0);
        __syncthreads();                            // buf0 reads done
        if (kk + 64 < CC) stage(kk + 64, 0);

        // ---- step k = kk+32: consume wtile[1] + bB -----------------------
        WAIT_ASYNC(2);                              // buf1 stage landed
        __syncthreads();
        if (kk + 64 < CC) loadB(bA, kk + 64);
        mma(bB, 1);
        __syncthreads();                            // buf1 reads done
        if (kk + 96 < CC) stage(kk + 96, 1);
    }

    const int grp = lane >> 4;
    const int cN  = lane & 15;
#pragma unroll
    for (int r = 0; r < 4; ++r) {
#pragma unroll
        for (int c = 0; c < 2; ++c) {
#pragma unroll
            for (int e = 0; e < 8; ++e) {
                int o = o0 + r * 16 + e + 8 * grp;        // C-tile row (ISA layout)
                int n = n0 + c * 16 + cN;                 // C-tile col
                float y = acc[r][c][e] * bnS[o] + bnT[o];
                if (MODE == 0) {
                    _Float16* out = (_Float16*)outp;      // [b][n][c]
                    out[((size_t)b * NN + n) * CC + o] = (_Float16)spikef(y);
                } else if (MODE == 1) {
                    _Float16* out = (_Float16*)outp;      // [b][c][n]
                    out[((size_t)b * CC + o) * NN + n] = (_Float16)spikef(y);
                } else {
                    float* out = (float*)outp;            // [b][c][n] fp32
                    out[((size_t)b * CC + o) * NN + n] = y;
                }
            }
        }
    }
}

// ---- K3: per (b,h): M'[d2][d1] = 0.25 * sum_n k[d1,n]*v[d2,n] ------------
// A = k head [64 x NN] row-major (ld=NN); B^T = v head rows (ld=NN).
// n-loop unrolled by 2 with ping-pong fragment sets (no copy rotation).
__global__ __launch_bounds__(128) void kv_kernel(const _Float16* __restrict__ kCN,
                                                 const _Float16* __restrict__ vCN,
                                                 _Float16* __restrict__ Mt) {
    const int lane = threadIdx.x & 31;
    const int wave = threadIdx.x >> 5;
    const int bh = blockIdx.x;
    const int b  = bh >> 3;
    const int hh = bh & 7;
    const _Float16* __restrict__ kb = kCN + ((size_t)b * CC + hh * HD) * NN + (size_t)(wave * 16) * NN;
    const _Float16* __restrict__ vb = vCN + ((size_t)b * CC + hh * HD) * NN;
    const int d1 = wave * 16;

    v8f acc[4] = {};
    v16h aA, aB, bA[4], bB[4];

    aA = load_frag_a(kb, NN, lane);
#pragma unroll
    for (int c = 0; c < 4; ++c)
        bA[c] = load_frag_bT(vb + (size_t)(c * 16) * NN, NN, lane);

    for (int n = 0; n < NN; n += 64) {
        // load n+32 set, compute n set
        aB = load_frag_a(kb + n + 32, NN, lane);
#pragma unroll
        for (int c = 0; c < 4; ++c)
            bB[c] = load_frag_bT(vb + (size_t)(c * 16) * NN + n + 32, NN, lane);
#pragma unroll
        for (int c = 0; c < 4; ++c)
            acc[c] = wmma16(aA, bA[c], acc[c]);
        // load n+64 set (if any), compute n+32 set
        if (n + 64 < NN) {
            aA = load_frag_a(kb + n + 64, NN, lane);
#pragma unroll
            for (int c = 0; c < 4; ++c)
                bA[c] = load_frag_bT(vb + (size_t)(c * 16) * NN + n + 64, NN, lane);
        }
#pragma unroll
        for (int c = 0; c < 4; ++c)
            acc[c] = wmma16(aB, bB[c], acc[c]);
    }

    _Float16* __restrict__ m = Mt + (size_t)bh * HD * HD;
    const int grp = lane >> 4, cN = lane & 15;
#pragma unroll
    for (int c = 0; c < 4; ++c)
#pragma unroll
        for (int e = 0; e < 8; ++e) {
            int d2  = c * 16 + cN;
            int dd1 = d1 + e + 8 * grp;
            m[d2 * HD + dd1] = (_Float16)(acc[c][e] * 0.25f);  // fold scale*2
        }
}

// ---- K4: o[n,d2] = spike( q[n,:] @ M' ), out transposed [b][n][c] --------
__global__ __launch_bounds__(256) void qm_kernel(const _Float16* __restrict__ qT,
                                                 const _Float16* __restrict__ Mt,
                                                 _Float16* __restrict__ oT) {
    const int lane = threadIdx.x & 31;
    const int wave = threadIdx.x >> 5;
    const int bh = blockIdx.y;
    const int b  = bh >> 3;
    const int hh = bh & 7;
    const int n0 = blockIdx.x * 128 + wave * 16;
    const _Float16* __restrict__ qb = qT + ((size_t)b * NN + n0) * CC + hh * HD;
    const _Float16* __restrict__ m  = Mt + (size_t)bh * HD * HD;

    v8f acc[4] = {};
#pragma unroll
    for (int k = 0; k < HD; k += 32) {
        v16h a = load_frag_a(qb + k, CC, lane);
#pragma unroll
        for (int c = 0; c < 4; ++c)
            acc[c] = wmma16(a, load_frag_bT(m + (size_t)(c * 16) * HD + k, HD, lane), acc[c]);
    }

    _Float16* __restrict__ ob = oT + ((size_t)b * NN + n0) * CC + hh * HD;
    const int grp = lane >> 4, cN = lane & 15;
#pragma unroll
    for (int c = 0; c < 4; ++c)
#pragma unroll
        for (int e = 0; e < 8; ++e) {
            int d2 = c * 16 + cN;
            int rn = e + 8 * grp;
            ob[(size_t)rn * CC + d2] = (_Float16)spikef(acc[c][e]);
        }
}

// ---- host side ------------------------------------------------------------
extern "C" void kernel_launch(void* const* d_in, const int* in_sizes, int n_in,
                              void* d_out, int out_size, void* d_ws, size_t ws_size,
                              hipStream_t stream) {
    (void)in_sizes; (void)n_in; (void)out_size; (void)ws_size;

    const float* x = (const float*)d_in[0];
    const float *w[4], *g[4], *bb_[4], *mm[4], *vv[4];
    for (int i = 0; i < 4; ++i) {               // order: q, k, v, p
        w[i]   = (const float*)d_in[1 + 5 * i + 0];
        g[i]   = (const float*)d_in[1 + 5 * i + 1];
        bb_[i] = (const float*)d_in[1 + 5 * i + 2];
        mm[i]  = (const float*)d_in[1 + 5 * i + 3];
        vv[i]  = (const float*)d_in[1 + 5 * i + 4];
    }

    char* wsb = (char*)d_ws;
    size_t off = 0;
    auto alloc = [&](size_t bytes) -> void* {
        off = (off + 255) & ~(size_t)255;
        void* p = wsb + off;
        off += bytes;
        return p;
    };
    _Float16* wh[4]; float* s[4]; float* t[4];
    for (int i = 0; i < 4; ++i) {
        wh[i] = (_Float16*)alloc((size_t)CC * CC * 2);
        s[i]  = (float*)alloc(CC * 4);
        t[i]  = (float*)alloc(CC * 4);
    }
    const size_t actBytes = (size_t)BB * NN * CC * 2;    // 8.4 MB each
    _Float16* xT  = (_Float16*)alloc(actBytes);
    _Float16* qT  = (_Float16*)alloc(actBytes);
    _Float16* kCN = (_Float16*)alloc(actBytes);
    _Float16* vCN = (_Float16*)alloc(actBytes);
    _Float16* Mt  = (_Float16*)alloc((size_t)BB * NHEADS * HD * HD * 2);
    _Float16* oT  = (_Float16*)alloc(actBytes);          // total ws ~45 MB

    for (int i = 0; i < 4; ++i) {
        wconv_kernel<<<(CC * CC + 255) / 256, 256, 0, stream>>>(w[i], wh[i], CC * CC);
        bnprep_kernel<<<2, 256, 0, stream>>>(g[i], bb_[i], mm[i], vv[i], s[i], t[i]);
    }

    spike_x_kernel<<<dim3(NN / 32, CC / 32, BB), dim3(32, 8), 0, stream>>>(x, xT);

    const dim3 gg(64, 4);                                 // 8192/128 cols x 512/128 rows
    gemm_bn_kernel<0><<<gg, 256, 0, stream>>>(wh[0], xT, s[0], t[0], (void*)qT);
    gemm_bn_kernel<1><<<gg, 256, 0, stream>>>(wh[1], xT, s[1], t[1], (void*)kCN);
    gemm_bn_kernel<1><<<gg, 256, 0, stream>>>(wh[2], xT, s[2], t[2], (void*)vCN);

    kv_kernel<<<BB * NHEADS, 128, 0, stream>>>(kCN, vCN, Mt);
    qm_kernel<<<dim3(NN / 128, BB * NHEADS), 256, 0, stream>>>(qT, Mt, oT);

    gemm_bn_kernel<2><<<gg, 256, 0, stream>>>(wh[3], oT, s[3], t[3], d_out);
}
